// Attention_22_22256520527996
// MI455X (gfx1250) — matrix operand order repaired and checked by reading, not compiled, not run on hardware
//
#include <hip/hip_runtime.h>
#include <hip/hip_bf16.h>

// MI455X / gfx1250, wave32. Bandwidth-bound XCiT channel attention.
// Pipeline:
//  k_pack_w  : qkv_w -> f16 WMMA B-fragments (per quadrant)
//  k_ln_qkv  : channel-LN + 1x1 conv (192x64 GEMM) via v_wmma_f32_16x16x32_f16,
//              writes qkv as f16; B-fragments staged to LDS via TDM
//  k_gram    : depthwise 3x3 fused on the fly; per-(qd,b,head,row) partial
//              Gram matrix S=sum_n q k^T and ||q||^2, ||k||^2 (deterministic)
//  k_attn_M  : reduce partials, softmax(8x8)/head, fold attn + proj_w into one
//              64x64 matrix M per (qd,b); emit M as f16 WMMA B-fragments
//  k_out     : dwconv(v) on the fly -> WMMA GEMM with M -> +proj_b, +grw*x -> out
//              M-fragments staged to LDS via Tensor Data Mover.

typedef __attribute__((ext_vector_type(16))) _Float16 v16h;
typedef __attribute__((ext_vector_type(8)))  float    v8f;
typedef __attribute__((ext_vector_type(4)))  unsigned u32x4;
typedef __attribute__((ext_vector_type(8)))  int      i32x8;
typedef __attribute__((ext_vector_type(4)))  int      i32x4;

#define QDIM 192
#define NPIX (192*192)
#define CH   64
#define CH3  192

#if defined(__has_builtin)
#if __has_builtin(__builtin_amdgcn_tensor_load_to_lds) && __has_builtin(__builtin_amdgcn_s_wait_tensorcnt)
#define HAVE_TDM 1
#endif
#endif

#define LDSP __attribute__((address_space(3)))

union HF16 { _Float16 h[16]; v16h v; };
union HO8  { _Float16 h[8];  uint4 u; };
union HP2  { _Float16 h[2];  unsigned u; };

// A-fragment K index for 16-bit 16x32 A (contiguous, matching B fragments):
// lanes 0-15: VGPR j -> K = 2j,2j+1 (0..15); lanes 16-31: +16.
__device__ __forceinline__ int kmapA(int j, int half) {
  return 2*j + 16*half;
}

// 1-D contiguous TDM copy of `ndw` dwords (ndw < 65536) global -> LDS.
// Issue from one wave; follow with s_wait_tensorcnt(0) + block barrier.
__device__ __forceinline__ void tdm_copy_dwords(const void* gsrc, void* lds_dst,
                                                unsigned ndw) {
#if defined(HAVE_TDM)
  unsigned lds_off = (unsigned)(unsigned long long)(LDSP char*)lds_dst;
  unsigned long long ga = (unsigned long long)gsrc;
  u32x4 g0 = { 1u, lds_off, (unsigned)ga,
               (unsigned)(((ga >> 32) & 0x1FFFFFFull) | (2u << 30)) };   // count=1, type=2
  i32x8 g1 = { (int)(2u << 16),        // data_size=4B, workgroup_mask=0
               (int)(ndw << 16),       // tensor_dim0
               (int)(1u << 16),        // tensor_dim1 = 1
               (int)(ndw << 16),       // tile_dim0
               1,                      // tile_dim1 = 1
               (int)ndw, 0, 0 };       // tensor_dim0_stride
  i32x4 z4 = { 0, 0, 0, 0 };
#if __clang_major__ >= 23
  i32x8 z8 = { 0, 0, 0, 0, 0, 0, 0, 0 };
  __builtin_amdgcn_tensor_load_to_lds(g0, g1, z4, z4, z8, 0);
#else
  __builtin_amdgcn_tensor_load_to_lds(g0, g1, z4, z4, 0);
#endif
  __builtin_amdgcn_s_wait_tensorcnt(0);
#endif
}

// ---------------- kernel 0: pack qkv_w into B-fragment layout ----------------
// B 32x16 16-bit layout used here (contiguous k): VGPR j, lanes 0-15: K=2j,2j+1, N=lane;
// lanes 16-31: K=16+2j,16+2j+1, N=lane&15.  frag idx = ((t*2+ks)*32+lane)*8+j
__global__ __launch_bounds__(256) void k_pack_w(const float* __restrict__ qkv_w,
                                                unsigned* __restrict__ bfrag) {
  int idx = blockIdx.x * 256 + threadIdx.x;            // 24576 total
  int qd = idx / 6144; int rem = idx % 6144;
  int t  = rem >> 9;  rem &= 511;
  int ks = rem >> 8;  rem &= 255;
  int ln = rem >> 3;  int j = rem & 7;
  int k0 = 2*j + ((ln >> 4) & 1) * 16 + ks * 32;
  int n  = t * 16 + (ln & 15);
  HP2 p;
  p.h[0] = (_Float16)qkv_w[(qd*CH3 + n)*CH + k0];
  p.h[1] = (_Float16)qkv_w[(qd*CH3 + n)*CH + k0 + 1];
  bfrag[idx] = p.u;
}

// ---------------- kernel 1: LayerNorm + QKV 1x1 conv (WMMA) -----------------
// wave = one 16-pixel tile (1 row x 16 cols) -> all 192 qkv channels.
// M=pixels(16), N=out channels(16/tile), K=64 (2 WMMA k-steps of 32).
__global__ __launch_bounds__(256) void k_ln_qkv(
    const float* __restrict__ x, const float* __restrict__ ln_w,
    const float* __restrict__ ln_b, const float* __restrict__ qkv_b,
    const unsigned* __restrict__ bfrag, _Float16* __restrict__ qkv_h) {
  __shared__ __align__(16) float xs[8][CH][16];
  __shared__ float mu_s[8][16], ex2_s[8][16];
  __shared__ float bs[CH3];                            // qkv_b staged
  __shared__ __align__(32) unsigned bf[6144];          // 12 tiles x 2 ks x 32 x 8

  int tid = threadIdx.x, lane = tid & 31, wave = tid >> 5;
  int qd = blockIdx.y;
  int t  = blockIdx.x * 8 + wave;                      // per-qd tile id
  int wt = t % 12, h = (t / 12) % QDIM, b = t / (12 * QDIM);
  int qy = qd >> 1, qx = qd & 1;
  int grow = qy*QDIM + h, gcol0 = qx*QDIM + wt*16;

#if defined(HAVE_TDM)
  if (tid < 32) tdm_copy_dwords(bfrag + qd * 6144, bf, 6144u);   // wave 0 DMA
#else
  {
    const uint4* src = (const uint4*)(bfrag + qd * 6144);
    uint4* dst = (uint4*)bf;
    for (int i = tid; i < 1536; i += 256) dst[i] = src[i];
  }
#endif
  if (tid < CH3) bs[tid] = qkv_b[qd*CH3 + tid];

  // x tile: 64 ch x 16 px -> LDS
  for (int c = lane; c < CH; c += 32) {
    const float4* xv = (const float4*)(x + ((size_t)(b*CH + c)*384 + grow)*384 + gcol0);
    float* d = &xs[wave][c][0];
    ((float4*)d)[0] = xv[0]; ((float4*)d)[1] = xv[1];
    ((float4*)d)[2] = xv[2]; ((float4*)d)[3] = xv[3];
  }
  __syncthreads();

  int m = lane & 15, half = lane >> 4;
  if (half == 0) {                                     // per-pixel mean
    float s = 0.f;
    for (int c = 0; c < CH; ++c) s += xs[wave][c][m];
    mu_s[wave][m] = s * (1.0f/CH);
  } else {                                             // per-pixel E[x^2]
    float s = 0.f;
    for (int c = 0; c < CH; ++c) { float v = xs[wave][c][m]; s += v*v; }
    ex2_s[wave][m] = s * (1.0f/CH);
  }
  __syncthreads();

  float mu   = mu_s[wave][m];
  float rstd = rsqrtf(ex2_s[wave][m] - mu*mu + 1e-5f);
  const float* lnw = ln_w + qd*CH;
  const float* lnb = ln_b + qd*CH;
  HF16 A[2];
  #pragma unroll
  for (int ks = 0; ks < 2; ++ks)
    #pragma unroll
    for (int j = 0; j < 8; ++j) {
      int k = kmapA(j, half) + 32*ks;
      A[ks].h[2*j]   = (_Float16)((xs[wave][k][m]   - mu)*rstd*lnw[k]   + lnb[k]);
      A[ks].h[2*j+1] = (_Float16)((xs[wave][k+1][m] - mu)*rstd*lnw[k+1] + lnb[k+1]);
    }

  // D layout: acc[j] = pixel (j + 8*half), channel n = tt*16 + (lane&15)
  _Float16* dst = qkv_h + ((size_t)(qd*4 + b)*CH3 + m)*NPIX
                        + (size_t)h*QDIM + wt*16 + 8*half;
  for (int tt = 0; tt < 12; ++tt) {
    v16h B0 = *(const v16h*)&bf[((tt*2 + 0)*32 + lane)*8];
    v16h B1 = *(const v16h*)&bf[((tt*2 + 1)*32 + lane)*8];
    v8f acc = {};
    acc = __builtin_amdgcn_wmma_f32_16x16x32_f16(false, A[0].v, false, B0, (short)0, acc, false, false);
    acc = __builtin_amdgcn_wmma_f32_16x16x32_f16(false, A[1].v, false, B1, (short)0, acc, false, false);
    float bias = bs[tt*16 + m];
    HO8 o;
    #pragma unroll
    for (int j = 0; j < 8; ++j) o.h[j] = (_Float16)(acc[j] + bias);
    *(uint4*)dst = o.u;                                // 8 consecutive f16 px
    dst += (size_t)16 * NPIX;
  }
}

// ---------------- kernel 2: Gram reduction (dwconv fused) -------------------
// block = (h-row, head, qd*4+b). Computes dwconv(q),dwconv(k) for one row,
// then partial S[8][8], ||q||^2[8], ||k||^2[8] -> 80 floats, deterministically.
__global__ __launch_bounds__(256) void k_gram(
    const _Float16* __restrict__ qkv_h, const float* __restrict__ dw_w,
    const float* __restrict__ dw_b, float* __restrict__ partial) {
  __shared__ _Float16 tile[16][3][194];                // 8 q-ch + 8 k-ch, rows h-1..h+1
  __shared__ float dwqk[QDIM][16];
  __shared__ float wls[16][9];
  __shared__ float bls[16];
  int tid = threadIdx.x;
  int h = blockIdx.x, head = blockIdx.y, qb = blockIdx.z;
  int qd = qb >> 2;
  const size_t base = (size_t)qb * CH3 * NPIX;

  if (tid < 144) {                                     // stage dw weights
    int c = tid / 9, e = tid % 9;
    int gch = (c < 8) ? head*8 + c : CH + head*8 + (c - 8);
    wls[c][e] = dw_w[(qd*CH3 + gch)*9 + e];
    if (e == 0) bls[c] = dw_b[qd*CH3 + gch];
  }
  for (int idx = tid; idx < 16*3*194; idx += 256) {
    int c = idx / (3*194); int rem = idx % (3*194);
    int r = rem / 194, cc = rem % 194;
    int gh = h + r - 1, gw = cc - 1;
    int gch = (c < 8) ? head*8 + c : CH + head*8 + (c - 8);
    _Float16 v = (_Float16)0.f;
    if (gh >= 0 && gh < QDIM && gw >= 0 && gw < QDIM)
      v = qkv_h[base + (size_t)gch*NPIX + gh*QDIM + gw];
    tile[c][r][cc] = v;
  }
  __syncthreads();

  if (tid < QDIM) {                                    // one column per thread
    #pragma unroll
    for (int c = 0; c < 16; ++c) {
      float s = bls[c];
      #pragma unroll
      for (int dy = 0; dy < 3; ++dy)
        #pragma unroll
        for (int dx = 0; dx < 3; ++dx)
          s += (float)tile[c][dy][tid + dx] * wls[c][dy*3 + dx];
      dwqk[tid][c] = s;
    }
  }
  __syncthreads();

  if (tid < 80) {                                      // deterministic row reduce
    float s = 0.f;
    if (tid < 64) {
      int c = tid >> 3, d = tid & 7;
      for (int w = 0; w < QDIM; ++w) s += dwqk[w][c] * dwqk[w][8 + d];
    } else if (tid < 72) {
      int c = tid - 64;
      for (int w = 0; w < QDIM; ++w) { float v = dwqk[w][c]; s += v*v; }
    } else {
      int d = tid - 72;
      for (int w = 0; w < QDIM; ++w) { float v = dwqk[w][8 + d]; s += v*v; }
    }
    partial[(((size_t)qb*8 + head)*QDIM + h)*80 + tid] = s;
  }
}

// ---------------- kernel 3: softmax + fold proj into M, emit M-fragments ----
__global__ __launch_bounds__(256) void k_attn_M(
    const float* __restrict__ partial, const float* __restrict__ temp,
    const float* __restrict__ proj_w, unsigned* __restrict__ mfrag) {
  __shared__ float S[8][80];
  __shared__ float attnW[8][8][8];
  __shared__ float Mm[64][64];
  int tid = threadIdx.x;
  int qb = blockIdx.x, qd = qb >> 2;

  for (int task = tid; task < 640; task += 256) {      // reduce 192 rows
    int head = task / 80, v = task % 80;
    const float* p = partial + ((size_t)qb*8 + head)*QDIM*80 + v;
    float s = 0.f;
    for (int r = 0; r < QDIM; ++r) s += p[r*80];
    S[head][v] = s;
  }
  __syncthreads();

  if (tid < 64) {                                      // softmax row per thread
    int head = tid >> 3, c = tid & 7;
    float tmp = temp[qd*8 + head];
    float qn = fmaxf(sqrtf(S[head][64 + c]), 1e-12f);
    float lg[8]; float mx = -3.4e38f;
    #pragma unroll
    for (int d = 0; d < 8; ++d) {
      float kn = fmaxf(sqrtf(S[head][72 + d]), 1e-12f);
      float l = S[head][c*8 + d] / (qn*kn) * tmp;
      lg[d] = l; mx = fmaxf(mx, l);
    }
    float se = 0.f;
    #pragma unroll
    for (int d = 0; d < 8; ++d) { lg[d] = __expf(lg[d] - mx); se += lg[d]; }
    float inv = 1.f / se;
    #pragma unroll
    for (int d = 0; d < 8; ++d) attnW[head][c][d] = lg[d]*inv;
  }
  __syncthreads();

  // M[o][d] = sum_{c in head(d)} proj_w[o][c] * attnW[head][c'][d']
  for (int idx = tid; idx < 4096; idx += 256) {
    int o = idx >> 6, d = idx & 63;
    int head = d >> 3, dd = d & 7;
    float s = 0.f;
    #pragma unroll
    for (int j = 0; j < 8; ++j)
      s += proj_w[(qd*CH + o)*CH + head*8 + j] * attnW[head][j][dd];
    Mm[o][d] = s;
  }
  __syncthreads();

  for (int idx = tid; idx < 2048; idx += 256) {        // B-fragment layout
    int t = idx >> 9; int rem = idx & 511;
    int ks = rem >> 8; rem &= 255;
    int ln = rem >> 3; int j = rem & 7;
    int k0 = 2*j + ((ln >> 4) & 1)*16 + ks*32;
    int n  = t*16 + (ln & 15);
    HP2 p;
    p.h[0] = (_Float16)Mm[n][k0];
    p.h[1] = (_Float16)Mm[n][k0 + 1];
    mfrag[qb*2048 + idx] = p.u;
  }
}

// ---------------- kernel 4: out = M @ dwconv(v) + proj_b + grw*x (WMMA) -----
__global__ __launch_bounds__(256) void k_out(
    const _Float16* __restrict__ qkv_h, const float* __restrict__ dw_w,
    const float* __restrict__ dw_b, const unsigned* __restrict__ mfrag,
    const float* __restrict__ proj_b, const float* __restrict__ grw,
    const float* __restrict__ x, float* __restrict__ out) {
  __shared__ _Float16 vs[8][CH][54];                   // [ch][r*18+cc] halo tile
  __shared__ __align__(32) unsigned mf[2048];          // M fragments (8 KB)
  __shared__ float wls[CH][9];                         // dwconv weights for v
  __shared__ float bls[CH];
  __shared__ float pbs[CH];                            // proj_b staged

  int tid = threadIdx.x, lane = tid & 31, wave = tid >> 5;
  int qd = blockIdx.y;
  int t  = blockIdx.x * 8 + wave;
  int wt = t % 12, h = (t / 12) % QDIM, b = t / (12 * QDIM);
  int b0 = (blockIdx.x * 8) / (12 * QDIM);             // uniform per block

#if defined(HAVE_TDM)
  if (tid < 32) tdm_copy_dwords(mfrag + (size_t)(qd*4 + b0)*2048, mf, 2048u);
#else
  for (int i = tid; i < 2048; i += 256) mf[i] = mfrag[(size_t)(qd*4 + b0)*2048 + i];
#endif

  // stage dwconv(v) weights + biases + proj_b
  for (int i = tid; i < CH*9; i += 256) {
    int c = i / 9, e = i % 9;
    wls[c][e] = dw_w[(qd*CH3 + 2*CH + c)*9 + e];
  }
  if (tid < CH) {
    bls[tid] = dw_b[qd*CH3 + 2*CH + tid];
    pbs[tid] = proj_b[qd*CH + tid];
  }

  // v tile with 3x3 halo: 64 ch x 3 rows x 18 cols (f16)
  size_t vbase = ((size_t)(qd*4 + b)*CH3 + 2*CH)*NPIX;
  for (int i = lane; i < CH*54; i += 32) {
    int c = i / 54, rem = i % 54;
    int r = rem / 18, cc = rem % 18;
    int gh = h + r - 1, gw = wt*16 + cc - 1;
    _Float16 v = (_Float16)0.f;
    if (gh >= 0 && gh < QDIM && gw >= 0 && gw < QDIM)
      v = qkv_h[vbase + (size_t)c*NPIX + gh*QDIM + gw];
    vs[wave][c][rem] = v;
  }
  __syncthreads();

  int m = lane & 15, half = lane >> 4;
  HF16 A[2];                                           // dwconv(v) fragments
  #pragma unroll
  for (int ks = 0; ks < 2; ++ks)
    #pragma unroll
    for (int j = 0; j < 8; ++j) {
      int k = kmapA(j, half) + 32*ks;
      #pragma unroll
      for (int e = 0; e < 2; ++e) {
        int kk = k + e;
        float s = bls[kk];
        #pragma unroll
        for (int dy = 0; dy < 3; ++dy)
          #pragma unroll
          for (int dx = 0; dx < 3; ++dx)
            s += (float)vs[wave][kk][dy*18 + m + dx] * wls[kk][dy*3 + dx];
        A[ks].h[2*j + e] = (_Float16)s;
      }
    }

  int qy = qd >> 1, qx = qd & 1;
  int grow = qy*QDIM + h, gcol = qx*QDIM + wt*16 + 8*half;
  float g = grw[qd];
  for (int tt = 0; tt < 4; ++tt) {
    v16h B0 = *(const v16h*)&mf[((tt*2 + 0)*32 + lane)*8];
    v16h B1 = *(const v16h*)&mf[((tt*2 + 1)*32 + lane)*8];
    v8f acc = {};
    acc = __builtin_amdgcn_wmma_f32_16x16x32_f16(false, A[0].v, false, B0, (short)0, acc, false, false);
    acc = __builtin_amdgcn_wmma_f32_16x16x32_f16(false, A[1].v, false, B1, (short)0, acc, false, false);
    int n = tt*16 + m;
    float pb = pbs[n];
    size_t addr = ((size_t)(b*CH + n)*384 + grow)*384 + gcol;
    const float4* xr = (const float4*)(x + addr);
    float4 r0 = xr[0], r1 = xr[1];
    float4 o0, o1;
    o0.x = g*r0.x + acc[0] + pb;  o0.y = g*r0.y + acc[1] + pb;
    o0.z = g*r0.z + acc[2] + pb;  o0.w = g*r0.w + acc[3] + pb;
    o1.x = g*r1.x + acc[4] + pb;  o1.y = g*r1.y + acc[5] + pb;
    o1.z = g*r1.z + acc[6] + pb;  o1.w = g*r1.w + acc[7] + pb;
    float4* op = (float4*)(out + addr);
    op[0] = o0; op[1] = o1;
  }
}

// ---------------------------------------------------------------------------
extern "C" void kernel_launch(void* const* d_in, const int* in_sizes, int n_in,
                              void* d_out, int out_size, void* d_ws, size_t ws_size,
                              hipStream_t stream) {
  const float* x      = (const float*)d_in[0];
  const float* ln_w   = (const float*)d_in[1];
  const float* ln_b   = (const float*)d_in[2];
  const float* qkv_w  = (const float*)d_in[3];
  const float* qkv_b  = (const float*)d_in[4];
  const float* dw_w   = (const float*)d_in[5];
  const float* dw_b   = (const float*)d_in[6];
  const float* temp   = (const float*)d_in[7];
  const float* proj_w = (const float*)d_in[8];
  const float* proj_b = (const float*)d_in[9];
  const float* grw    = (const float*)d_in[10];
  float* out = (float*)d_out;

  // workspace layout (bytes): qkv f16 216MiB | B-frags | Gram partials | M-frags
  char* ws = (char*)d_ws;
  const size_t off_qkv  = 0;
  const size_t off_bf   = off_qkv + (size_t)16 * CH3 * NPIX * 2;   // 226,492,416
  const size_t off_part = off_bf + 98304;
  const size_t off_mf   = off_part + (size_t)16 * 8 * QDIM * 80 * 4;
  _Float16* qkv_h   = (_Float16*)(ws + off_qkv);
  unsigned* bfrag   = (unsigned*)(ws + off_bf);
  float*    partial = (float*)   (ws + off_part);
  unsigned* mfrag   = (unsigned*)(ws + off_mf);

  k_pack_w <<<96, 256, 0, stream>>>(qkv_w, bfrag);
  k_ln_qkv <<<dim3(1152, 4), 256, 0, stream>>>(x, ln_w, ln_b, qkv_b, bfrag, qkv_h);
  k_gram   <<<dim3(QDIM, 8, 16), 256, 0, stream>>>(qkv_h, dw_w, dw_b, partial);
  k_attn_M <<<16, 256, 0, stream>>>(partial, temp, proj_w, mfrag);
  k_out    <<<dim3(1152, 4), 256, 0, stream>>>(qkv_h, dw_w, dw_b, mfrag, proj_b, grw, x, out);
}
